// SpikeAttention_86397562126871
// MI455X (gfx1250) — matrix operand order repaired and checked by reading, not compiled, not run on hardware
//
#include <hip/hip_runtime.h>

// ---------------- problem constants ----------------
#define BB 2
#define TT 4
#define NN 512
#define DD 768
#define HH 12
#define DH 64
#define BT (BB*TT)            // 8
#define MTOT (BT*NN)          // 4096 rows of x
#define NX (MTOT*DD)          // 3145728 elements of x / out
#define NW (DD*DD)            // 589824 per weight
#define NSPK (BT*HH*NN*NN)    // 25165824 spike elements
#define RATE_IDX NX           // d_out[RATE_IDX] = attn_rate

// workspace element offsets (ushort = bf16 storage)
#define O_XBF 0u
#define O_WQ  3145728u
#define O_WK  3735552u
#define O_WV  4325376u
#define O_WO  4915200u
#define O_QB  5505024u          // (bth, n, dh)     96*512*64
#define O_KB  8650752u          // (bth, n, dh)
#define O_VT  11796480u         // (bth, dh, n)  transposed V
#define O_SPK 14942208u         // (bth, n, m)      96*512*512
#define O_CTX 40108032u         // (m, h*64+dh)     4096*768
// total = 43253760 elems = 86.5 MB

typedef __attribute__((ext_vector_type(16))) __bf16          v16bf;
typedef __attribute__((ext_vector_type(8)))  float           v8f;
typedef __attribute__((ext_vector_type(8)))  unsigned short  v8us;
typedef __attribute__((ext_vector_type(16))) unsigned short  v16us;

__device__ __forceinline__ unsigned short f2bf(float f) {
    unsigned u = __builtin_bit_cast(unsigned, f);
    u += 0x7FFFu + ((u >> 16) & 1u);          // round-to-nearest-even
    return (unsigned short)(u >> 16);
}

__device__ __forceinline__ v8f wmma_bf16(v16bf a, v16bf b, v8f c) {
    return __builtin_amdgcn_wmma_f32_16x16x32_bf16(
        /*neg_a=*/false, a, /*neg_b=*/false, b,
        /*c_mod=*/(short)0, c, /*reuse_a=*/false, /*reuse_b=*/false);
}

// A-fragment 16x32 bf16 (ISA 7.12.2): lanes 0-15 -> M=lane, K pairs 0..7 then 16..23;
// lanes 16-31 -> K offset +8.  Two 16B loads per lane.
__device__ __forceinline__ v16bf load_a_frag(const unsigned short* p, int row, int ld,
                                             int kk, int lane) {
    int r   = row + (lane & 15);
    int klo = kk + ((lane < 16) ? 0 : 8);
    const unsigned short* q = p + (size_t)r * ld + klo;
    v16us u;
    *(v8us*)&u       = *(const v8us*)q;          // elems 0..7  : K klo..klo+7
    *(((v8us*)&u)+1) = *(const v8us*)(q + 16);   // elems 8..15 : K klo+16..+23
    return __builtin_bit_cast(v16bf, u);
}

// B-fragment 32x16 bf16: lane = column (mod 16); lanes 0-15 hold K kk..kk+7 then kk+16..kk+23,
// lanes 16-31 hold K offset +8, packed 2/VGPR.  Source matrix stored as
// [col][K] row-major (i.e. the operand's transpose), so two 16B loads per lane.
__device__ __forceinline__ v16bf load_b_frag(const unsigned short* p, int col, int ld,
                                             int kk, int lane) {
    int c  = col + (lane & 15);
    int kb = kk + ((lane < 16) ? 0 : 8);
    const unsigned short* q = p + (size_t)c * ld + kb;
    v16us u;
    *(v8us*)&u       = *(const v8us*)q;
    *(((v8us*)&u)+1) = *(const v8us*)(q + 16);
    return __builtin_bit_cast(v16bf, u);
}

// ---- 32x32 output tile GEMM core: C += A[arow:+32, :] @ Bt[bcol:+32, :]^T ----
// Register double-buffered over K; acc order: [a0b0, a0b1, a1b0, a1b1].
template <int KSTEPS>
__device__ __forceinline__ void gemm32(const unsigned short* A, int arow, int lda,
                                       const unsigned short* Bt, int bcol, int ldb,
                                       int lane, v8f acc[4]) {
    v16bf a0 = load_a_frag(A, arow,      lda, 0, lane);
    v16bf a1 = load_a_frag(A, arow + 16, lda, 0, lane);
    v16bf b0 = load_b_frag(Bt, bcol,      ldb, 0, lane);
    v16bf b1 = load_b_frag(Bt, bcol + 16, ldb, 0, lane);
#pragma unroll
    for (int s = 0; s < KSTEPS; ++s) {
        v16bf na0 = a0, na1 = a1, nb0 = b0, nb1 = b1;
        if (s + 1 < KSTEPS) {                 // prefetch next K-step into regs
            int kk = (s + 1) * 32;
            na0 = load_a_frag(A, arow,      lda, kk, lane);
            na1 = load_a_frag(A, arow + 16, lda, kk, lane);
            nb0 = load_b_frag(Bt, bcol,      ldb, kk, lane);
            nb1 = load_b_frag(Bt, bcol + 16, ldb, kk, lane);
            if (s + 2 < KSTEPS)
                __builtin_prefetch(A + (size_t)(arow + (lane & 15)) * lda + kk + 32, 0, 3);
        }
        acc[0] = wmma_bf16(a0, b0, acc[0]);
        acc[1] = wmma_bf16(a0, b1, acc[1]);
        acc[2] = wmma_bf16(a1, b0, acc[2]);
        acc[3] = wmma_bf16(a1, b1, acc[3]);
        a0 = na0; a1 = na1; b0 = nb0; b1 = nb1;
    }
}

// ---------------- kernel 1: cast inputs to bf16, zero rate ----------------
__global__ void cast_kernel(const float* x, const float* wq, const float* wk,
                            const float* wv, const float* wo,
                            unsigned short* ws, float* out) {
    int tid = blockIdx.x * blockDim.x + threadIdx.x;
    if (tid == 0) out[RATE_IDX] = 0.0f;
    if (tid < NX) {
        ws[O_XBF + tid] = f2bf(x[tid]);
    } else if (tid < NX + 4 * NW) {
        int j = tid - NX;
        int w = j / NW, r = j % NW;
        const float* src = (w == 0) ? wq : (w == 1) ? wk : (w == 2) ? wv : wo;
        ws[O_WQ + (unsigned)w * NW + r] = f2bf(src[r]);
    }
}

// ---------------- kernel 2: projection  y = x @ W^T  (32x32/wave) ----------
// MODE 0: store q head-layout   (bth, n, dh)
// MODE 1: store k head-layout   (bth, n, dh)
// MODE 2: store v transposed    (bth, dh, n)
template <int MODE>
__global__ void proj_kernel(const unsigned short* ws_in, unsigned short* ws_out,
                            unsigned wofs, unsigned dofs) {
    const unsigned short* xb = ws_in + O_XBF;
    const unsigned short* wb = ws_in + wofs;
    unsigned short* dst = ws_out + dofs;

    int lane = threadIdx.x & 31;
    int widx = blockIdx.x * 4 + (threadIdx.x >> 5);   // 3072 waves
    int ct = widx % 24;                                // 768/32 col tiles
    int mt = widx / 24;                                // 4096/32 row tiles
    int mrow = mt * 32, c0 = ct * 32;

    v8f acc[4] = {v8f{}, v8f{}, v8f{}, v8f{}};
    gemm32<24>(xb, mrow, DD, wb, c0, DD, lane, acc);

#pragma unroll
    for (int q = 0; q < 4; ++q) {
        int c = c0 + (q & 1) * 16 + (lane & 15);
        int h = c >> 6, dh = c & 63;
#pragma unroll
        for (int v = 0; v < 8; ++v) {
            int m = mrow + (q >> 1) * 16 + v + ((lane < 16) ? 0 : 8);
            int bt = m >> 9, n = m & 511;
            int bth = bt * HH + h;
            unsigned short val = f2bf(acc[q][v]);
            if (MODE == 2)
                dst[((size_t)bth * DH + dh) * NN + n] = val;   // transposed V
            else
                dst[((size_t)bth * NN + n) * DH + dh] = val;
        }
    }
}

// ---------------- kernel 3: scores (Q@K^T) + fused LIF over T --------------
__global__ void score_lif_kernel(const unsigned short* ws_in, unsigned short* ws_out,
                                 float* out) {
    const unsigned short* qb = ws_in + O_QB;
    const unsigned short* kb = ws_in + O_KB;
    unsigned short* spk = ws_out + O_SPK;

    int lane = threadIdx.x & 31;
    int widx = blockIdx.x * 4 + (threadIdx.x >> 5);   // 6144 waves
    int bh  = widx / 256;                              // 24 (b,h) pairs
    int rem = widx % 256;
    int it = rem >> 4, jt = rem & 15;                  // 16x16 tiles of 32x32
    int b = bh / HH, h = bh % HH;

    const float beta  = 0.951229424500714f;            // exp(-1/20)
    const float scale = 0.125f;                        // 64^-0.5
    v8f mem[4] = {v8f{}, v8f{}, v8f{}, v8f{}};
    float lsum = 0.0f;

    for (int t = 0; t < TT; ++t) {
        int bth = (b * TT + t) * HH + h;
        const unsigned short* qp = qb + (size_t)bth * NN * DH;
        const unsigned short* kp = kb + (size_t)bth * NN * DH;
        v8f s[4] = {v8f{}, v8f{}, v8f{}, v8f{}};
        gemm32<2>(qp, it * 32, DH, kp, jt * 32, DH, lane, s);

        unsigned short* sp = spk + (size_t)bth * NN * NN;
#pragma unroll
        for (int q = 0; q < 4; ++q) {
            int mm = jt * 32 + (q & 1) * 16 + (lane & 15);
#pragma unroll
            for (int v = 0; v < 8; ++v) {
                mem[q][v] = beta * mem[q][v] + s[q][v] * scale;  // LIF integrate
                float sv = (mem[q][v] >= 0.5f) ? 1.0f : 0.0f;
                mem[q][v] -= sv * 0.5f;                          // soft reset
                lsum += sv;
                int n = it * 32 + (q >> 1) * 16 + v + ((lane < 16) ? 0 : 8);
                sp[(size_t)n * NN + mm] = (sv != 0.0f) ? (unsigned short)0x3F80u
                                                       : (unsigned short)0u;
            }
        }
    }
    // block reduce spike count -> attn_rate
    __shared__ float red[128];
    red[threadIdx.x] = lsum;
    __syncthreads();
    for (int s = 64; s > 0; s >>= 1) {
        if (threadIdx.x < s) red[threadIdx.x] += red[threadIdx.x + s];
        __syncthreads();
    }
    if (threadIdx.x == 0)
        atomicAdd(out + RATE_IDX, red[0] * (1.0f / (float)NSPK));
}

// ---------------- kernel 4: spikes @ V  -> context (bf16) ------------------
__global__ void av_kernel(const unsigned short* ws_in, unsigned short* ws_out) {
    const unsigned short* spk = ws_in + O_SPK;
    const unsigned short* vt  = ws_in + O_VT;
    unsigned short* ctx = ws_out + O_CTX;

    int lane = threadIdx.x & 31;
    int widx = blockIdx.x * 4 + (threadIdx.x >> 5);   // 3072 waves
    int bth = widx / 32;                               // 96
    int rem = widx % 32;
    int nt = rem >> 1, dt = rem & 1;                   // 16 row tiles x 2 dh tiles

    const unsigned short* sp = spk + (size_t)bth * NN * NN;
    const unsigned short* vp = vt  + (size_t)bth * DH * NN;

    v8f acc[4] = {v8f{}, v8f{}, v8f{}, v8f{}};
    gemm32<16>(sp, nt * 32, NN, vp, dt * 32, NN, lane, acc);

    int bt = bth / HH, h = bth % HH;
#pragma unroll
    for (int q = 0; q < 4; ++q) {
        int dh = dt * 32 + (q & 1) * 16 + (lane & 15);
#pragma unroll
        for (int v = 0; v < 8; ++v) {
            int n = nt * 32 + (q >> 1) * 16 + v + ((lane < 16) ? 0 : 8);
            ctx[((size_t)(bt * NN + n)) * DD + h * DH + dh] = f2bf(acc[q][v]);
        }
    }
}

// ---------------- kernel 5: output projection -> d_out (fp32) --------------
__global__ void outproj_kernel(const unsigned short* ws_in, float* out) {
    const unsigned short* ctx = ws_in + O_CTX;
    const unsigned short* wo  = ws_in + O_WO;

    int lane = threadIdx.x & 31;
    int widx = blockIdx.x * 4 + (threadIdx.x >> 5);   // 3072 waves
    int ct = widx % 24;
    int mt = widx / 24;
    int mrow = mt * 32, c0 = ct * 32;

    v8f acc[4] = {v8f{}, v8f{}, v8f{}, v8f{}};
    gemm32<24>(ctx, mrow, DD, wo, c0, DD, lane, acc);

#pragma unroll
    for (int q = 0; q < 4; ++q) {
        int c = c0 + (q & 1) * 16 + (lane & 15);
#pragma unroll
        for (int v = 0; v < 8; ++v) {
            int m = mrow + (q >> 1) * 16 + v + ((lane < 16) ? 0 : 8);
            out[(size_t)m * DD + c] = acc[q][v];
        }
    }
}

extern "C" void kernel_launch(void* const* d_in, const int* in_sizes, int n_in,
                              void* d_out, int out_size, void* d_ws, size_t ws_size,
                              hipStream_t stream) {
    const float* x  = (const float*)d_in[0];
    const float* wq = (const float*)d_in[1];
    const float* wk = (const float*)d_in[2];
    const float* wv = (const float*)d_in[3];
    const float* wo = (const float*)d_in[4];
    float* out = (float*)d_out;
    unsigned short* ws = (unsigned short*)d_ws;

    // 1) bf16 casts + zero rate accumulator
    {
        int total = NX + 4 * NW;
        cast_kernel<<<(total + 255) / 256, 256, 0, stream>>>(x, wq, wk, wv, wo, ws, out);
    }
    // 2) Q / K / V^T projections: 3072 wave-tiles each, 4 waves per block
    proj_kernel<0><<<768, 128, 0, stream>>>(ws, ws, O_WQ, O_QB);
    proj_kernel<1><<<768, 128, 0, stream>>>(ws, ws, O_WK, O_KB);
    proj_kernel<2><<<768, 128, 0, stream>>>(ws, ws, O_WV, O_VT);
    // 3) scores + LIF: 6144 wave-tiles
    score_lif_kernel<<<1536, 128, 0, stream>>>(ws, ws, out);
    // 4) spikes @ V: 3072 wave-tiles
    av_kernel<<<768, 128, 0, stream>>>(ws, ws);
    // 5) output projection: 3072 wave-tiles
    outproj_kernel<<<768, 128, 0, stream>>>(ws, out);
}
